// Model_82832739271075
// MI455X (gfx1250) — matrix operand order repaired and checked by reading, not compiled, not run on hardware
//
#include <hip/hip_runtime.h>
#include <math.h>

// ============================================================================
// Spherical CNN (s2cnn) pipeline for MI455X / gfx1250.
// Every contraction (band-limited DFTs, Wigner contractions, kernel-spectrum
// products, inverse SO(3) transforms) is lowered onto ONE generic batched
// stride-generalized GEMM microkernel built on v_wmma_f32_16x16x32_f16
// (wave32; each wave computes a 16x32 C tile = 2 WMMA issues per K-step,
// f16 inputs / f32 accumulation).  Full tiles take a uniform fast path with
// unconditional strided loads + global_prefetch of the next K tile; edge
// tiles use the guarded path.  All constant tables (Wigner-d via closed-form
// log-factorial sum in fp64, twiddles, quadrature weights) are generated
// on-device each call (graph-capture safe).
// bias1/bias2 are omitted: they cancel exactly inside the following BatchNorm.
// Requires ws_size >= ~126M floats (~504 MB).
// ============================================================================

typedef __attribute__((ext_vector_type(16))) _Float16 v16h;
typedef __attribute__((ext_vector_type(8)))  float    v8f;

#define PI_D 3.14159265358979323846

// ---------------- workspace offsets (in floats) ----------------
// persistent tables
static const long O_W1RE=0, O_W1IM=3968;            // [a=128][m=31] e^{-2pi i m~ a/128}
static const long O_V1RE=7936, O_V1IM=8928;         // [n=31][b=32]  e^{+2pi i n~ b/32}
static const long O_W2RE=9920, O_W2IM=10528;        // [b=32][j=19]  e^{-2pi i j~ b/32}
static const long O_V2RE=11136, O_V2IM=11516;       // [n=19][b=20]  e^{+2pi i n~ b/20}
static const long O_QW1=11896, O_QW2=12024, O_QW3=12056;
static const long O_BN1M=12076, O_BN1R=12176, O_BN2M=12276, O_BN2R=12376;
static const long O_FEAT=12476;
static const long O_DC1=13276;     // [m=31][b=128][l=16] d^l_{m~,0}(beta1_b)*qw1[b]
static const long O_G1B=76764;     // [m=31][n=31][l=16][q=32] (2l+1) d^l_{m~,n~}(beta2_q) d^l_{n~,0}(pi/2)
static const long O_DW2T=568796;   // [m=19][p=19][q=32][l=10] d^l_{m~,p~}(beta2_q)*qw2[q]
static const long O_DP2T=684316;   // [l=10][p=19][n=19] d^l_{n~,p~}(pi/2)
static const long O_DI2T=687926;   // [m=19][n=19][l=10][q=20] (2l+1) d^l_{m~,n~}(beta3_q)
// dynamic buffers (lifetime-overlapped regions)
static const long O_XMRE=800000,  O_XMIM=990464;      // Xm [m][zf=48][b=128]
static const long O_XHRE=1180928, O_XHIM=1204736;     // xh [m][zf][l=16]
static const long O_XHPRE=1228544,O_XHPIM=1252352;    // xhp [(z,l,m)=3968][f=6]
static const long O_KMRE=1276160, O_KMIM=1294760;     // km [n=31][f=6][g=100]
static const long O_TRE=1313360,  O_TIM=13614160;     // t  [n][(z,l,m)=3968][g=100]
static const long O_T2RE=25914960,O_T2IM=38215760;    // t2 [(m,n)=961][zg=800][l=16]
static const long O_SRE=800000,   O_SIM=50516560;     // S  [(m,n)][zg][q=32]
static const long O_URE=75118160, O_UIM=100513360;    // U  [m][(zg,q)=25600][b=32]
static const long O_Z1=800000;                        // z1 [(zg,q)=25600][a=32][b=32]
static const long O_PRE=27014400, O_PIM=42579200;     // P  [(zf,q)=25600][a=32][j=19]
static const long O_XMNRE=58144000,O_XMNIM=67385600;  // Xmn[(zf,q)][i=19][j=19]
static const long O_XH2RE=800000, O_XH2IM=3688000;    // xh2[l=10][(zf,m)=15200][p=19]
static const long O_VRE=6576000,  O_VIM=9464000;      // v  [l][(zf,m)][n=19]
static const long O_VPRE=12352000,O_VPIM=15240000;    // vp [n=19][(z,l,m)=1520][f=100]
static const long O_KM2RE=18128000,O_KM2IM=18318000;  // km2[n=19][f=100][g=100]
static const long O_ZHRE=18508000,O_ZHIM=21396000;    // zh [n][(z,l,m)][g=100]
static const long O_ZHPRE=24284000,O_ZHPIM=27172000;  // zhp[(m,n)=361][zg=800][l=10]
static const long O_S2RE=30060000,O_S2IM=35836000;    // S2 [(m,n)][zg][q=20]
static const long O_S2PRE=41612000,O_S2PIM=47388000;  // S2p[m=19][(zg,q)=16000][n=19]
static const long O_U2RE=53164000,O_U2IM=59244000;    // U2 [m][(zg,q)][b=20]
static const long O_Z2=65324000;                      // z2 [(zg,q)=16000][a=20][b=20]

// ============================================================================
// Generic batched, fully-strided GEMM:  C[b] (+)= alpha * A[b] @ B[b]
// addr(A) = bsA*b + rsA*m + csA*k ; addr(B) = bsB*b + rsB*k + csB*n ;
// addr(C) = bsC*b + rsC*m + csC*n.  One wave (32 threads) per 16x32 C tile
// (two 16x16 WMMA sub-tiles sharing one A fragment).
// ============================================================================
struct GemmP {
  const float* A; const float* B; float* C;
  int M, N, K;
  long bsA, rsA, csA, bsB, rsB, csB, bsC, rsC, csC;
  float alpha; int beta;
};

__global__ __launch_bounds__(32) void k_gemm(GemmP p) {
  __shared__ _Float16 la[16 * 32];   // [row 16][k 32]
  __shared__ _Float16 lb[32 * 32];   // [k 32][col 32]
  const int lane = threadIdx.x;
  const int tn = blockIdx.x * 32;
  const int tm = blockIdx.y * 16;
  const long bat = blockIdx.z;
  const float* Ab = p.A + bat * p.bsA;
  const float* Bb = p.B + bat * p.bsB;
  float* Cb = p.C + bat * p.bsC;
  const int hi = lane >> 4, lr = lane & 15;
  const bool fullM = (tm + 16 <= p.M);
  const bool fullN = (tn + 32 <= p.N);

  v8f c0, c1;
  if (p.beta) {
#pragma unroll
    for (int r = 0; r < 8; ++r) {
      int row = tm + r + hi * 8, col = tn + lr;
      c0[r] = (row < p.M && col < p.N) ? Cb[(long)row * p.rsC + (long)col * p.csC] : 0.0f;
      c1[r] = (row < p.M && (col + 16) < p.N)
                  ? Cb[(long)row * p.rsC + (long)(col + 16) * p.csC] : 0.0f;
    }
  } else {
#pragma unroll
    for (int r = 0; r < 8; ++r) { c0[r] = 0.0f; c1[r] = 0.0f; }
  }

  for (int k0 = 0; k0 < p.K; k0 += 32) {
    const bool fullK = (k0 + 32 <= p.K);
    // ---- A tile 16x32 into LDS (f16, alpha folded) ----
    if (fullM && fullK) {                 // uniform fast path, no predication
      const int r = lr, kb = hi * 16;
      const float* src = Ab + (long)(tm + r) * p.rsA + (long)(k0 + kb) * p.csA;
      float t[16];
#pragma unroll
      for (int k = 0; k < 16; ++k) t[k] = src[(long)k * p.csA];
      __builtin_prefetch(src + 32 * p.csA, 0, 1);   // next K tile
#pragma unroll
      for (int k = 0; k < 16; ++k) la[r * 32 + kb + k] = (_Float16)(p.alpha * t[k]);
    } else {
      for (int i = lane; i < 512; i += 32) {
        int r = i >> 5, k = i & 31;
        float x = 0.0f;
        if ((tm + r) < p.M && (k0 + k) < p.K)
          x = p.alpha * Ab[(long)(tm + r) * p.rsA + (long)(k0 + k) * p.csA];
        la[i] = (_Float16)x;
      }
    }
    // ---- B tile 32x32 into LDS (f16) ----
    if (fullN && fullK) {                 // uniform fast path: lane = k row
      const float* src = Bb + (long)(k0 + lane) * p.rsB + (long)tn * p.csB;
      float t[32];
#pragma unroll
      for (int n = 0; n < 32; ++n) t[n] = src[(long)n * p.csB];
      __builtin_prefetch(src + 32 * p.rsB, 0, 1);   // next K tile
#pragma unroll
      for (int n = 0; n < 32; ++n) lb[lane * 32 + n] = (_Float16)t[n];
    } else {
      for (int i = lane; i < 1024; i += 32) {
        int k = i >> 5, n = i & 31;
        float x = 0.0f;
        if ((k0 + k) < p.K && (tn + n) < p.N)
          x = Bb[(long)(k0 + k) * p.rsB + (long)(tn + n) * p.csB];
        lb[i] = (_Float16)x;
      }
    }
    __syncthreads();
    // Build WMMA fragments per CDNA5 wave32 register layouts (ISA 7.12.2).
    v16h af, bf0, bf1;
#pragma unroll
    for (int i = 0; i < 16; ++i) {
      int vv = i >> 1, pb = i & 1;
      int ka = (vv < 4 ? 2 * vv + pb : 16 + 2 * (vv - 4) + pb) + (hi ? 8 : 0);
      af[i] = la[lr * 32 + ka];
      int kb = (vv < 4 ? 2 * vv + pb : 16 + 2 * (vv - 4) + pb) + (hi ? 8 : 0);
      bf0[i] = lb[kb * 32 + lr];
      bf1[i] = lb[kb * 32 + 16 + lr];
    }
    c0 = __builtin_amdgcn_wmma_f32_16x16x32_f16(false, af, false, bf0, (short)0, c0,
                                                false, false);
    c1 = __builtin_amdgcn_wmma_f32_16x16x32_f16(false, af, false, bf1, (short)0, c1,
                                                false, false);
    __syncthreads();
  }
#pragma unroll
  for (int r = 0; r < 8; ++r) {
    int row = tm + r + hi * 8, col = tn + lr;
    if (row < p.M && col < p.N) Cb[(long)row * p.rsC + (long)col * p.csC] = c0[r];
    if (row < p.M && (col + 16) < p.N)
      Cb[(long)row * p.rsC + (long)(col + 16) * p.csC] = c1[r];
  }
}

// ---------------- generic 5D gather-permute: out contiguous ----------------
__global__ void k_perm(const float* in, float* out, long total,
                       int d1, int d2, int d3, int d4,
                       long s0, long s1, long s2, long s3, long s4) {
  long i = (long)blockIdx.x * 256 + threadIdx.x;
  if (i >= total) return;
  long r = i;
  int i4 = (int)(r % d4); r /= d4;
  int i3 = (int)(r % d3); r /= d3;
  int i2 = (int)(r % d2); r /= d2;
  int i1 = (int)(r % d1); r /= d1;
  out[i] = in[r * s0 + (long)i1 * s1 + (long)i2 * s2 + (long)i3 * s3 + (long)i4 * s4];
}

// ============================================================================
// Constant-table generation (fp64 on device)
// ============================================================================
__device__ double wig_d(int l, int mp, int mm, double beta) {
  // d^l_{mp,mm}(beta) = <mp| exp(-i beta Jy) |mm>  == reference's exp(-beta A/2)
  if (mp < -l || mp > l || mm < -l || mm > l) return 0.0;
  double lf[32];
  lf[0] = 0.0;
  for (int n = 1; n < 32; ++n) lf[n] = lf[n - 1] + log((double)n);
  double ch = cos(0.5 * beta), sh = sin(0.5 * beta);
  int smin = (mm - mp) > 0 ? (mm - mp) : 0;
  int smax = (l + mm) < (l - mp) ? (l + mm) : (l - mp);
  double pref = 0.5 * (lf[l + mp] + lf[l - mp] + lf[l + mm] + lf[l - mm]);
  double acc = 0.0;
  for (int s = smin; s <= smax; ++s) {
    double lt = pref - lf[l + mm - s] - lf[s] - lf[mp - mm + s] - lf[l - mp - s];
    double t = exp(lt) * pow(ch, (double)(2 * l + mm - mp - 2 * s)) *
               pow(sh, (double)(mp - mm + 2 * s));
    acc += ((mp - mm + s) & 1) ? -t : t;
  }
  return acc;
}

__device__ double quadw(int j, int b) {
  double beta = PI_D * (2 * j + 1) / (4.0 * b);
  double s = 0.0;
  for (int k = 0; k < b; ++k) s += sin(beta * (2 * k + 1)) / (double)(2 * k + 1);
  return 2.0 / b * sin(beta) * s;
}

__global__ void k_small(float* W) {
  long g = (long)blockIdx.x * blockDim.x + threadIdx.x;
  long st = (long)gridDim.x * blockDim.x;
  for (long i = g; i < 128 * 31; i += st) {         // W1 [a][m], e^{-i th}
    long a = i / 31, m = i % 31;
    double th = 2.0 * PI_D * (double)(m - 15) * (double)a / 128.0;
    W[O_W1RE + i] = (float)cos(th); W[O_W1IM + i] = (float)(-sin(th));
  }
  for (long i = g; i < 31 * 32; i += st) {          // V1 [n][b], e^{+i th}
    long n = i / 32, b = i % 32;
    double th = 2.0 * PI_D * (double)(n - 15) * (double)b / 32.0;
    W[O_V1RE + i] = (float)cos(th); W[O_V1IM + i] = (float)sin(th);
  }
  for (long i = g; i < 32 * 19; i += st) {          // W2 [b][j], e^{-i th}
    long b = i / 19, j = i % 19;
    double th = 2.0 * PI_D * (double)(j - 9) * (double)b / 32.0;
    W[O_W2RE + i] = (float)cos(th); W[O_W2IM + i] = (float)(-sin(th));
  }
  for (long i = g; i < 19 * 20; i += st) {          // V2 [n][b], e^{+i th}
    long n = i / 20, b = i % 20;
    double th = 2.0 * PI_D * (double)(n - 9) * (double)b / 20.0;
    W[O_V2RE + i] = (float)cos(th); W[O_V2IM + i] = (float)sin(th);
  }
  for (long i = g; i < 128; i += st) W[O_QW1 + i] = (float)quadw((int)i, 64);
  for (long i = g; i < 32;  i += st) W[O_QW2 + i] = (float)quadw((int)i, 16);
  for (long i = g; i < 20;  i += st) W[O_QW3 + i] = (float)quadw((int)i, 10);
}

__global__ void k_dc1(float* W) {
  long i = (long)blockIdx.x * 256 + threadIdx.x;
  if (i >= 31L * 128 * 16) return;
  int m = (int)(i / (128 * 16)), b = (int)((i / 16) % 128), l = (int)(i % 16);
  double bet = PI_D * (2 * b + 1) / 256.0;
  W[O_DC1 + i] = (float)(wig_d(l, m - 15, 0, bet) * (double)W[O_QW1 + b]);
}
__global__ void k_g1b(float* W) {
  long i = (long)blockIdx.x * 256 + threadIdx.x;
  if (i >= 31L * 31 * 16 * 32) return;
  int m = (int)(i / (31 * 16 * 32)), n = (int)((i / (16 * 32)) % 31);
  int l = (int)((i / 32) % 16), q = (int)(i % 32);
  double bet = PI_D * (2 * q + 1) / 64.0;
  W[O_G1B + i] = (float)((2 * l + 1) * wig_d(l, m - 15, n - 15, bet) *
                         wig_d(l, n - 15, 0, 0.5 * PI_D));
}
__global__ void k_dw2t(float* W) {
  long i = (long)blockIdx.x * 256 + threadIdx.x;
  if (i >= 19L * 19 * 32 * 10) return;
  int m = (int)(i / (19 * 32 * 10)), p = (int)((i / 320) % 19);
  int q = (int)((i / 10) % 32), l = (int)(i % 10);
  double bet = PI_D * (2 * q + 1) / 64.0;
  W[O_DW2T + i] = (float)(wig_d(l, m - 9, p - 9, bet) * (double)W[O_QW2 + q]);
}
__global__ void k_dp2t(float* W) {
  long i = (long)blockIdx.x * 256 + threadIdx.x;
  if (i >= 10L * 19 * 19) return;
  int l = (int)(i / 361), p = (int)((i / 19) % 19), n = (int)(i % 19);
  W[O_DP2T + i] = (float)wig_d(l, n - 9, p - 9, 0.5 * PI_D);
}
__global__ void k_di2t(float* W) {
  long i = (long)blockIdx.x * 256 + threadIdx.x;
  if (i >= 19L * 19 * 10 * 20) return;
  int m = (int)(i / (19 * 10 * 20)), n = (int)((i / 200) % 19);
  int l = (int)((i / 20) % 10), q = (int)(i % 20);
  double bet = PI_D * (2 * q + 1) / 40.0;
  W[O_DI2T + i] = (float)((2 * l + 1) * wig_d(l, m - 9, n - 9, bet));
}

// ============================================================================
// BatchNorm (per-channel over z,q,a,b) + ReLU, pooling, classifier head
// ============================================================================
__global__ __launch_bounds__(256) void k_bnstats(const float* x, float* mo, float* ro,
                                                 int nz, long zs, long gs, long nper) {
  __shared__ double ss[256], sq[256];
  int g = blockIdx.x, t = threadIdx.x;
  double s = 0.0, q = 0.0;
  for (int z = 0; z < nz; ++z) {
    const float* p = x + (long)z * zs + (long)g * gs;
    for (long e = t; e < nper; e += 256) { double v = p[e]; s += v; q += v * v; }
  }
  ss[t] = s; sq[t] = q; __syncthreads();
  for (int o = 128; o > 0; o >>= 1) {
    if (t < o) { ss[t] += ss[t + o]; sq[t] += sq[t + o]; }
    __syncthreads();
  }
  if (t == 0) {
    double N = (double)nz * (double)nper;
    double mu = ss[0] / N, var = sq[0] / N - mu * mu;
    mo[g] = (float)mu; ro[g] = (float)(1.0 / sqrt(var + 1e-5));
  }
}
__global__ void k_bnapply(float* x, const float* mo, const float* ro,
                          const float* gam, const float* bet,
                          long total, long gdiv, int gmod) {
  long i = (long)blockIdx.x * 256 + threadIdx.x;
  if (i >= total) return;
  int g = (int)((i / gdiv) % gmod);
  float v = (x[i] - mo[g]) * ro[g] * gam[g] + bet[g];
  x[i] = v > 0.0f ? v : 0.0f;
}
__global__ __launch_bounds__(256) void k_feat(const float* z2, const float* qw3, float* feat) {
  __shared__ float ss[256];
  int b = blockIdx.x, t = threadIdx.x;
  const float* p = z2 + (long)b * 8000;
  float s = 0.0f;
  for (int e = t; e < 8000; e += 256) s += qw3[e / 400] * p[e];
  ss[t] = s; __syncthreads();
  for (int o = 128; o > 0; o >>= 1) { if (t < o) ss[t] += ss[t + o]; __syncthreads(); }
  if (t == 0) feat[b] = ss[0];
}
__global__ __launch_bounds__(128) void k_head(const float* feat, const float* wout,
                                              const float* bout, float* out) {
  __shared__ float lg[80];
  int t = threadIdx.x;
  if (t < 80) {
    int z = t / 10, c = t % 10;
    float s = bout[c];
    for (int f = 0; f < 100; ++f) s += feat[z * 100 + f] * wout[c * 100 + f];
    lg[t] = s;
  }
  __syncthreads();
  if (t < 8) {
    float m = -1e30f;
    for (int c = 0; c < 10; ++c) m = fmaxf(m, lg[t * 10 + c]);
    float se = 0.0f;
    for (int c = 0; c < 10; ++c) se += expf(lg[t * 10 + c] - m);
    float lse = logf(se);
    for (int c = 0; c < 10; ++c) out[t * 10 + c] = lg[t * 10 + c] - m - lse;
  }
}

// ============================================================================
// Host orchestration
// ============================================================================
static inline void gemm(hipStream_t s, const float* A, const float* B, float* C,
                        int M, int N, int K, int batch,
                        long bsA, long rsA, long csA,
                        long bsB, long rsB, long csB,
                        long bsC, long rsC, long csC, float alpha, int beta) {
  GemmP p{A, B, C, M, N, K, bsA, rsA, csA, bsB, rsB, csB, bsC, rsC, csC, alpha, beta};
  dim3 grid((N + 31) / 32, (M + 15) / 16, batch);
  k_gemm<<<grid, dim3(32), 0, s>>>(p);
}
static inline void perm(hipStream_t s, const float* in, float* out, long total,
                        int d1, int d2, int d3, int d4,
                        long s0, long s1, long s2, long s3, long s4) {
  k_perm<<<(unsigned)((total + 255) / 256), 256, 0, s>>>(in, out, total, d1, d2, d3, d4,
                                                         s0, s1, s2, s3, s4);
}

extern "C" void kernel_launch(void* const* d_in, const int* in_sizes, int n_in,
                              void* d_out, int out_size, void* d_ws, size_t ws_size,
                              hipStream_t stream) {
  (void)in_sizes; (void)n_in; (void)out_size; (void)ws_size;
  const float* x    = (const float*)d_in[0];   // (8,6,128,128)
  const float* k1   = (const float*)d_in[1];   // (6,100,128)
  const float* gam1 = (const float*)d_in[3];
  const float* bet1 = (const float*)d_in[4];
  const float* k2   = (const float*)d_in[5];   // (100,100,32)
  const float* gam2 = (const float*)d_in[7];
  const float* bet2 = (const float*)d_in[8];
  const float* wout = (const float*)d_in[9];   // (10,100)
  const float* bout = (const float*)d_in[10];  // (10)
  float* out = (float*)d_out;                  // (8,10)
  float* W = (float*)d_ws;
  hipStream_t s = stream;

  const float S1 = (float)(1.0 / sqrt(12288.0));   // 1/sqrt(2*B1*F0*B2^4/B1^2)
  const float S2 = (float)(1.0 / sqrt(781.25));    // 1/sqrt(2*B2*F1*B3^3/B2^3)

  // ---- constant tables ----
  k_small<<<64, 256, 0, s>>>(W);
  k_dc1 <<<(31 * 128 * 16 + 255) / 256, 256, 0, s>>>(W);
  k_g1b <<<(31 * 31 * 16 * 32 + 255) / 256, 256, 0, s>>>(W);
  k_dw2t<<<(19 * 19 * 32 * 10 + 255) / 256, 256, 0, s>>>(W);
  k_dp2t<<<(10 * 19 * 19 + 255) / 256, 256, 0, s>>>(W);
  k_di2t<<<(19 * 19 * 10 * 20 + 255) / 256, 256, 0, s>>>(W);

  // ================= Stage A: s2_conv =================
  // A1: band-limited DFT of x: Xm[m][zf][b] = sum_a x[zf,b,a] e^{-2pi i m~ a/128}
  gemm(s, x, W + O_W1RE, W + O_XMRE, 6144, 31, 128, 1, 0, 128, 1, 0, 31, 1, 0, 1, 6144, 1.0f, 0);
  gemm(s, x, W + O_W1IM, W + O_XMIM, 6144, 31, 128, 1, 0, 128, 1, 0, 31, 1, 0, 1, 6144, 1.0f, 0);
  // A2: xh[m][zf][l] = sum_b Xm * DC1[m][b][l]
  gemm(s, W + O_XMRE, W + O_DC1, W + O_XHRE, 48, 16, 128, 31, 6144, 128, 1, 2048, 16, 1, 768, 16, 1, 1.0f, 0);
  gemm(s, W + O_XMIM, W + O_DC1, W + O_XHIM, 48, 16, 128, 31, 6144, 128, 1, 2048, 16, 1, 768, 16, 1, 1.0f, 0);
  // repack xh[m][(z,f)][l] -> xhp[(z,l,m)][f]
  perm(s, W + O_XHRE, W + O_XHPRE, 23808, 16, 31, 6, 1, 96, 1, 768, 16, 0);
  perm(s, W + O_XHIM, W + O_XHPIM, 23808, 16, 31, 6, 1, 96, 1, 768, 16, 0);
  // A3: km[n][f][g] = S1 * sum_j k1[f,g,j] e^{-i n~ a_j}
  gemm(s, k1, W + O_W1RE, W + O_KMRE, 600, 31, 128, 1, 0, 128, 1, 0, 31, 1, 0, 1, 600, S1, 0);
  gemm(s, k1, W + O_W1IM, W + O_KMIM, 600, 31, 128, 1, 0, 128, 1, 0, 31, 1, 0, 1, 600, S1, 0);
  // A4: t[n][(z,l,m)][g] = sum_f xh[z,f,l,m] * conj(km[n,f,g])
  gemm(s, W + O_XHPRE, W + O_KMRE, W + O_TRE, 3968, 100, 6, 31, 0, 6, 1, 600, 100, 1, 396800, 100, 1,  1.0f, 0);
  gemm(s, W + O_XHPIM, W + O_KMIM, W + O_TRE, 3968, 100, 6, 31, 0, 6, 1, 600, 100, 1, 396800, 100, 1,  1.0f, 1);
  gemm(s, W + O_XHPIM, W + O_KMRE, W + O_TIM, 3968, 100, 6, 31, 0, 6, 1, 600, 100, 1, 396800, 100, 1,  1.0f, 0);
  gemm(s, W + O_XHPRE, W + O_KMIM, W + O_TIM, 3968, 100, 6, 31, 0, 6, 1, 600, 100, 1, 396800, 100, 1, -1.0f, 1);
  // repack t -> t2[(m,n)][zg][l]
  perm(s, W + O_TRE, W + O_T2RE, 12300800, 31, 8, 100, 16, 100, 396800, 49600, 1, 3100);
  perm(s, W + O_TIM, W + O_T2IM, 12300800, 31, 8, 100, 16, 100, 396800, 49600, 1, 3100);
  // A5: S[(m,n)][zg][q] = sum_l t2 * G1B[(m,n)][l][q]   (DI1 * DPI2_1 folded)
  gemm(s, W + O_T2RE, W + O_G1B, W + O_SRE, 800, 32, 16, 961, 12800, 16, 1, 512, 32, 1, 25600, 32, 1, 1.0f, 0);
  gemm(s, W + O_T2IM, W + O_G1B, W + O_SIM, 800, 32, 16, 961, 12800, 16, 1, 512, 32, 1, 25600, 32, 1, 1.0f, 0);
  // A6 pass1 (inverse DFT over n): U[m][(zg,q)][b] = sum_n S * V1[n][b]
  gemm(s, W + O_SRE, W + O_V1RE, W + O_URE, 25600, 32, 31, 31, 793600, 1, 25600, 0, 32, 1, 819200, 32, 1,  1.0f, 0);
  gemm(s, W + O_SIM, W + O_V1IM, W + O_URE, 25600, 32, 31, 31, 793600, 1, 25600, 0, 32, 1, 819200, 32, 1, -1.0f, 1);
  gemm(s, W + O_SRE, W + O_V1IM, W + O_UIM, 25600, 32, 31, 31, 793600, 1, 25600, 0, 32, 1, 819200, 32, 1,  1.0f, 0);
  gemm(s, W + O_SIM, W + O_V1RE, W + O_UIM, 25600, 32, 31, 31, 793600, 1, 25600, 0, 32, 1, 819200, 32, 1,  1.0f, 1);
  // A6 pass2 (Re of inverse DFT over m): z1 = cos@Ur - sin@Ui
  gemm(s, W + O_V1RE, W + O_URE, W + O_Z1, 32, 32, 31, 25600, 0, 1, 32, 32, 819200, 1, 1024, 32, 1,  1.0f, 0);
  gemm(s, W + O_V1IM, W + O_UIM, W + O_Z1, 32, 32, 31, 25600, 0, 1, 32, 32, 819200, 1, 1024, 32, 1, -1.0f, 1);
  // BN1 + ReLU (bias1 cancels inside BN)
  k_bnstats<<<100, 256, 0, s>>>(W + O_Z1, W + O_BN1M, W + O_BN1R, 8, 3276800, 32768, 32768);
  k_bnapply<<<(26214400 + 255) / 256, 256, 0, s>>>(W + O_Z1, W + O_BN1M, W + O_BN1R, gam1, bet1, 26214400, 32768, 100);

  // ================= Stage C: so3_conv =================
  // C1 pass1: P[(zf,q)][a][j] = sum_b z1[a,b] e^{-2pi i j~ b/32}
  gemm(s, W + O_Z1, W + O_W2RE, W + O_PRE, 32, 19, 32, 25600, 1024, 32, 1, 0, 19, 1, 608, 19, 1, 1.0f, 0);
  gemm(s, W + O_Z1, W + O_W2IM, W + O_PIM, 32, 19, 32, 25600, 1024, 32, 1, 0, 19, 1, 608, 19, 1, 1.0f, 0);
  // C1 pass2: Xmn[(zf,q)][i][j] = sum_a e^{-2pi i i~ a/32} P[a][j]
  gemm(s, W + O_W2RE, W + O_PRE, W + O_XMNRE, 19, 19, 32, 25600, 0, 1, 19, 608, 19, 1, 361, 19, 1,  1.0f, 0);
  gemm(s, W + O_W2IM, W + O_PIM, W + O_XMNRE, 19, 19, 32, 25600, 0, 1, 19, 608, 19, 1, 361, 19, 1, -1.0f, 1);
  gemm(s, W + O_W2RE, W + O_PIM, W + O_XMNIM, 19, 19, 32, 25600, 0, 1, 19, 608, 19, 1, 361, 19, 1,  1.0f, 0);
  gemm(s, W + O_W2IM, W + O_PRE, W + O_XMNIM, 19, 19, 32, 25600, 0, 1, 19, 608, 19, 1, 361, 19, 1,  1.0f, 1);
  // C2: xh2[l][(zf,m)][p] = sum_q DW2[l,q,m,p] * Xmn[zf,q,m,p]
  gemm(s, W + O_XMNRE, W + O_DW2T, W + O_XH2RE, 800, 10, 32, 361, 1, 11552, 361, 320, 10, 1, 1, 361, 288800, 1.0f, 0);
  gemm(s, W + O_XMNIM, W + O_DW2T, W + O_XH2IM, 800, 10, 32, 361, 1, 11552, 361, 320, 10, 1, 1, 361, 288800, 1.0f, 0);
  // C3: km2[n][f][g] = S2 * sum_j k2[f,g,j] e^{-i n~ a_j}
  gemm(s, k2, W + O_W2RE, W + O_KM2RE, 10000, 19, 32, 1, 0, 32, 1, 0, 19, 1, 0, 1, 10000, S2, 0);
  gemm(s, k2, W + O_W2IM, W + O_KM2IM, 10000, 19, 32, 1, 0, 32, 1, 0, 19, 1, 0, 1, 10000, S2, 0);
  // C5a: v[l][(zf,m)][n] = sum_p xh2 * DP2T[l][p][n]
  gemm(s, W + O_XH2RE, W + O_DP2T, W + O_VRE, 15200, 19, 19, 10, 288800, 19, 1, 361, 19, 1, 288800, 19, 1, 1.0f, 0);
  gemm(s, W + O_XH2IM, W + O_DP2T, W + O_VIM, 15200, 19, 19, 10, 288800, 19, 1, 361, 19, 1, 288800, 19, 1, 1.0f, 0);
  // repack v[l][(z,f,m)][n] -> vp[n][(z,l,m)][f]
  perm(s, W + O_VRE, W + O_VPRE, 2888000, 8, 10, 19, 100, 1, 36100, 288800, 19, 361);
  perm(s, W + O_VIM, W + O_VPIM, 2888000, 8, 10, 19, 100, 1, 36100, 288800, 19, 361);
  // C5b: zh[n][(z,l,m)][g] = sum_f vp * conj(km2[n,f,g])
  gemm(s, W + O_VPRE, W + O_KM2RE, W + O_ZHRE, 1520, 100, 100, 19, 152000, 100, 1, 10000, 100, 1, 152000, 100, 1,  1.0f, 0);
  gemm(s, W + O_VPIM, W + O_KM2IM, W + O_ZHRE, 1520, 100, 100, 19, 152000, 100, 1, 10000, 100, 1, 152000, 100, 1,  1.0f, 1);
  gemm(s, W + O_VPIM, W + O_KM2RE, W + O_ZHIM, 1520, 100, 100, 19, 152000, 100, 1, 10000, 100, 1, 152000, 100, 1,  1.0f, 0);
  gemm(s, W + O_VPRE, W + O_KM2IM, W + O_ZHIM, 1520, 100, 100, 19, 152000, 100, 1, 10000, 100, 1, 152000, 100, 1, -1.0f, 1);
  // repack zh -> zhp[(m,n)][zg][l]
  perm(s, W + O_ZHRE, W + O_ZHPRE, 2888000, 19, 8, 100, 10, 100, 152000, 19000, 1, 1900);
  perm(s, W + O_ZHIM, W + O_ZHPIM, 2888000, 19, 8, 100, 10, 100, 152000, 19000, 1, 1900);
  // C6: S2[(m,n)][zg][q] = sum_l DI2[l,q,m,n] * zhp
  gemm(s, W + O_ZHPRE, W + O_DI2T, W + O_S2RE, 800, 20, 10, 361, 8000, 10, 1, 200, 20, 1, 16000, 20, 1, 1.0f, 0);
  gemm(s, W + O_ZHPIM, W + O_DI2T, W + O_S2IM, 800, 20, 10, 361, 8000, 10, 1, 200, 20, 1, 16000, 20, 1, 1.0f, 0);
  // repack S2 -> S2p[m][(zg,q)][n]
  perm(s, W + O_S2RE, W + O_S2PRE, 5776000, 8, 100, 20, 19, 304000, 2000, 20, 1, 16000);
  perm(s, W + O_S2IM, W + O_S2PIM, 5776000, 8, 100, 20, 19, 304000, 2000, 20, 1, 16000);
  // C7 pass1: U2[m][(zg,q)][b] = sum_n S2p * V2[n][b]
  gemm(s, W + O_S2PRE, W + O_V2RE, W + O_U2RE, 16000, 20, 19, 19, 304000, 19, 1, 0, 20, 1, 320000, 20, 1,  1.0f, 0);
  gemm(s, W + O_S2PIM, W + O_V2IM, W + O_U2RE, 16000, 20, 19, 19, 304000, 19, 1, 0, 20, 1, 320000, 20, 1, -1.0f, 1);
  gemm(s, W + O_S2PRE, W + O_V2IM, W + O_U2IM, 16000, 20, 19, 19, 304000, 19, 1, 0, 20, 1, 320000, 20, 1,  1.0f, 0);
  gemm(s, W + O_S2PIM, W + O_V2RE, W + O_U2IM, 16000, 20, 19, 19, 304000, 19, 1, 0, 20, 1, 320000, 20, 1,  1.0f, 1);
  // C7 pass2: z2 = cos@U2r - sin@U2i
  gemm(s, W + O_V2RE, W + O_U2RE, W + O_Z2, 20, 20, 19, 16000, 0, 1, 20, 20, 320000, 1, 400, 20, 1,  1.0f, 0);
  gemm(s, W + O_V2IM, W + O_U2IM, W + O_Z2, 20, 20, 19, 16000, 0, 1, 20, 20, 320000, 1, 400, 20, 1, -1.0f, 1);
  // BN2 + ReLU
  k_bnstats<<<100, 256, 0, s>>>(W + O_Z2, W + O_BN2M, W + O_BN2R, 8, 800000, 8000, 8000);
  k_bnapply<<<(6400000 + 255) / 256, 256, 0, s>>>(W + O_Z2, W + O_BN2M, W + O_BN2R, gam2, bet2, 6400000, 8000, 100);

  // ================= Head =================
  k_feat<<<800, 256, 0, s>>>(W + O_Z2, W + O_QW3, W + O_FEAT);
  k_head<<<1, 128, 0, s>>>(W + O_FEAT, wout, bout, out);
}
